// SingleHeadAttention_13597866459507
// MI455X (gfx1250) — matrix operand order repaired and checked by reading, not compiled, not run on hardware
//
#include <hip/hip_runtime.h>

typedef __bf16 bf16;
typedef __attribute__((ext_vector_type(16))) __bf16 v16bf;
typedef __attribute__((ext_vector_type(8)))  __bf16 v8bf;
typedef __attribute__((ext_vector_type(8)))  float  v8f;

#define BATCH 4
#define SEQ   2048
#define HDIM  1024
#define EDIM  1024
#define MTOT  (BATCH * SEQ)   // 8192 rows
#define SPAD  36              // LDS row padding (floats): 16B-aligned, conflict-free

// ---------------------------------------------------------------------------
// helpers
// ---------------------------------------------------------------------------
__device__ __forceinline__ v16bf load_frag16(const bf16* p0, const bf16* p1) {
    v8bf lo = *(const v8bf*)p0;
    v8bf hi = *(const v8bf*)p1;
    v16bf r;
#pragma unroll
    for (int i = 0; i < 8; ++i) { r[i] = lo[i]; r[8 + i] = hi[i]; }
    return r;
}

__device__ __forceinline__ float bcast_lane(float v, int srcLane) {
    int iv = __builtin_bit_cast(int, v);
    int r  = __builtin_amdgcn_ds_bpermute(srcLane << 2, iv);
    return __builtin_bit_cast(float, r);
}

// ---------------------------------------------------------------------------
// fp32 -> bf16 conversion
// ---------------------------------------------------------------------------
__global__ void cvt_f32_bf16(const float* __restrict__ in, bf16* __restrict__ out, int n) {
    int i = blockIdx.x * blockDim.x + threadIdx.x;
    int stride = gridDim.x * blockDim.x;
    for (; i < n; i += stride) out[i] = (bf16)in[i];
}

// ---------------------------------------------------------------------------
// Unified projection GEMM (NT):
//   out[a_row * ldo + b_row] = (sum_k A[a_row,k]*B[b_row,k] + bias[.]) * scale
// BIAS_A selects whether bias indexes A-rows (V^T case) or B-rows (Q/K case).
// Block: 256 thr = 8 waves = 2(M) x 4(N); block tile 64(M) x 256(N);
// each wave: 32x64 tile, 2x4 WMMA accumulators (8 wmmas per 6 fragment loads).
// ---------------------------------------------------------------------------
template <bool BIAS_A>
__global__ __launch_bounds__(256)
void proj_gemm(const bf16* __restrict__ A, const bf16* __restrict__ Bm,
               const float* __restrict__ bias, bf16* __restrict__ out,
               int ldo, float scale) {
    const int lane = threadIdx.x & 31;
    const int wid  = threadIdx.x >> 5;
    const int g    = lane >> 4;     // half-wave group (K-packing select)
    const int ln   = lane & 15;     // row / col within 16
    const int wm   = wid >> 2;      // 0..1
    const int wn   = wid & 3;       // 0..3
    const int Mbase = blockIdx.y * 64  + wm * 32;
    const int Nbase = blockIdx.x * 256 + wn * 64;

    v8f acc[2][4] = {};

    for (int kk = 0; kk < HDIM; kk += 32) {
        v16bf a[2], b[4];
#pragma unroll
        for (int i = 0; i < 2; ++i) {
            const bf16* p = A + (size_t)(Mbase + 16 * i + ln) * HDIM + kk + 16 * g;
            a[i] = load_frag16(p, p + 8);
        }
#pragma unroll
        for (int j = 0; j < 4; ++j) {
            const bf16* p = Bm + (size_t)(Nbase + 16 * j + ln) * HDIM + kk + 16 * g;
            b[j] = load_frag16(p, p + 8);
        }
#pragma unroll
        for (int i = 0; i < 2; ++i)
#pragma unroll
            for (int j = 0; j < 4; ++j)
                acc[i][j] = __builtin_amdgcn_wmma_f32_16x16x32_bf16(
                    false, a[i], false, b[j], (short)0, acc[i][j], false, false);
    }

    float bm[2][8];
    if (BIAS_A) {
#pragma unroll
        for (int i = 0; i < 2; ++i)
#pragma unroll
            for (int r = 0; r < 8; ++r)
                bm[i][r] = bias[Mbase + 16 * i + r + 8 * g];
    }
#pragma unroll
    for (int j = 0; j < 4; ++j) {
        const int n = Nbase + 16 * j + ln;
        const float bn = BIAS_A ? 0.f : bias[n];
#pragma unroll
        for (int i = 0; i < 2; ++i) {
#pragma unroll
            for (int r = 0; r < 8; ++r) {
                const int m = Mbase + 16 * i + r + 8 * g;
                const float bv = BIAS_A ? bm[i][r] : bn;
                out[(size_t)m * ldo + n] = (bf16)((acc[i][j][r] + bv) * scale);
            }
        }
    }
}

// ---------------------------------------------------------------------------
// One 32-key attention tile. MASKED=true only for the single diagonal tile.
// ---------------------------------------------------------------------------
template <bool MASKED>
__device__ __forceinline__ void attn_tile(
    int kb, int q_idx, int bS, int Ebase, int g, int ln, int wid,
    const bf16* __restrict__ K, const bf16* __restrict__ Vt,
    const v16bf (&qf)[4], v8f (&o)[8],
    float& m_prev, float& l_sum,
    float (*Sw)[16][SPAD], float (&Sred)[16][SPAD]) {

    // --- speculative prefetch of next tile's K/V fragment rows (no counters)
    if (kb + 32 < SEQ) {
        const int kn = kb + 32;
        __builtin_prefetch(K  + (size_t)(bS + kn + ln) * EDIM + Ebase + 16 * g, 0, 1);
        __builtin_prefetch(K  + (size_t)(bS + kn + 16 + ln) * EDIM + Ebase + 16 * g, 0, 1);
        __builtin_prefetch(Vt + (size_t)(Ebase + 16 * (wid & 7) + ln) * MTOT + bS + kn + 16 * g, 0, 1);
    }

    // --- load all 8 K fragments
    v16bf kf[8];
#pragma unroll
    for (int f = 0; f < 4; ++f) {
        const bf16* p0 = K + (size_t)(bS + kb + ln) * EDIM + Ebase + 32 * f + 16 * g;
        kf[f] = load_frag16(p0, p0 + 8);
        const bf16* p1 = K + (size_t)(bS + kb + 16 + ln) * EDIM + Ebase + 32 * f + 16 * g;
        kf[4 + f] = load_frag16(p1, p1 + 8);
    }

    // --- partial S = Q_slice * K_slice^T (two 16x16 tiles: keys kb..kb+31)
    v8f s0 = {}, s1 = {};
#pragma unroll
    for (int f = 0; f < 4; ++f) {
        s0 = __builtin_amdgcn_wmma_f32_16x16x32_bf16(false, qf[f], false, kf[f],     (short)0, s0, false, false);
        s1 = __builtin_amdgcn_wmma_f32_16x16x32_bf16(false, qf[f], false, kf[4 + f], (short)0, s1, false, false);
    }
#pragma unroll
    for (int r = 0; r < 8; ++r) {
        Sw[wid][r + 8 * g][ln]      = s0[r];
        Sw[wid][r + 8 * g][16 + ln] = s1[r];
    }
    __syncthreads();

    // --- cross-wave reduction: 512 elements over 256 threads
    for (int e = threadIdx.x; e < 512; e += 256) {
        const int rr = e >> 5, cc = e & 31;
        float s = 0.f;
#pragma unroll
        for (int w = 0; w < 8; ++w) s += Sw[w][rr][cc];
        Sred[rr][cc] = s;
    }
    __syncthreads();

    // --- pull this lane's score row into registers (8 x ds_load_b128)
    float srow[32];
    {
        const float4* rp = (const float4*)(&Sred[ln][0]);
#pragma unroll
        for (int c4 = 0; c4 < 8; ++c4) {
            const float4 v = rp[c4];
            srow[4 * c4 + 0] = v.x;
            srow[4 * c4 + 1] = v.y;
            srow[4 * c4 + 2] = v.z;
            srow[4 * c4 + 3] = v.w;
        }
    }

    // --- prefetch V fragments; latency hides under the softmax math below
    v16bf vf[8];
#pragma unroll
    for (int t = 0; t < 8; ++t) {
        const bf16* p = Vt + (size_t)(Ebase + 16 * t + ln) * MTOT + bS + kb + 16 * g;
        vf[t] = load_frag16(p, p + 8);
    }
    __syncthreads();   // Sw/Sred reads complete; LDS free for next tile

    // --- causal mask only on the diagonal tile
    if (MASKED) {
#pragma unroll
        for (int c = 0; c < 32; ++c)
            srow[c] = ((kb + c) <= q_idx) ? srow[c] : -1e30f;
    }

    // --- online softmax: max, exp in place, sum
    float m_new = m_prev;
#pragma unroll
    for (int c = 0; c < 32; ++c) m_new = fmaxf(m_new, srow[c]);
#pragma unroll
    for (int c = 0; c < 32; ++c) srow[c] = __expf(srow[c] - m_new);
    float psum = 0.f;
#pragma unroll
    for (int c = 0; c < 32; ++c) psum += srow[c];
    const float rescale = __expf(m_prev - m_new);
    l_sum = l_sum * rescale + psum;
    m_prev = m_new;

    // --- build P fragment (16x32 bf16, A-operand layout) from the exp row
    v16bf pf;
#pragma unroll
    for (int e = 0; e < 8; ++e) {
        pf[e]     = (bf16)srow[16 * g + e];
        pf[8 + e] = (bf16)srow[16 * g + 8 + e];
    }

    // --- rescale O accumulators (per-row factors via ds_bpermute)
    float fac[8];
#pragma unroll
    for (int r = 0; r < 8; ++r) fac[r] = bcast_lane(rescale, r + 8 * g);
#pragma unroll
    for (int t = 0; t < 8; ++t)
#pragma unroll
        for (int r = 0; r < 8; ++r) o[t][r] *= fac[r];

    // --- O += P * V_tile (fragments already in flight/registers)
#pragma unroll
    for (int t = 0; t < 8; ++t)
        o[t] = __builtin_amdgcn_wmma_f32_16x16x32_bf16(false, pf, false, vf[t], (short)0, o[t], false, false);
}

// ---------------------------------------------------------------------------
// Flash attention (causal). One block = 16 query rows of one batch.
// 8 waves; wave w owns E-slice [128w, 128w+128) of the output accumulator.
// All key tiles except the final diagonal tile run the unmasked fast path.
// ---------------------------------------------------------------------------
__global__ __launch_bounds__(256)
void flash_attn(const bf16* __restrict__ Q, const bf16* __restrict__ K,
                const bf16* __restrict__ Vt, float* __restrict__ out) {
    const int lane = threadIdx.x & 31;
    const int wid  = threadIdx.x >> 5;
    const int g    = lane >> 4;
    const int ln   = lane & 15;
    const int b    = blockIdx.y;
    const int q0   = blockIdx.x * 16;          // query base within batch
    const int bS   = b * SEQ;
    const int rowQ = bS + q0 + ln;             // global query row for frags
    const int Ebase = wid * 128;

    __shared__ alignas(16) float Sw[8][16][SPAD];   // per-wave partial scores
    __shared__ alignas(16) float Sred[16][SPAD];    // reduced scores

    // Preload Q fragments for this wave's E slice (kept in VGPRs all loop).
    v16bf qf[4];
#pragma unroll
    for (int f = 0; f < 4; ++f) {
        const bf16* p = Q + (size_t)rowQ * EDIM + Ebase + 32 * f + 16 * g;
        qf[f] = load_frag16(p, p + 8);
    }

    v8f o[8] = {};
    const int q_idx = q0 + ln;
    float m_prev = -1e30f, l_sum = 0.f;

    const int ktiles = (q0 + 15) / 32 + 1;
    for (int kt = 0; kt < ktiles - 1; ++kt)
        attn_tile<false>(kt * 32, q_idx, bS, Ebase, g, ln, wid, K, Vt,
                         qf, o, m_prev, l_sum, Sw, Sred);
    attn_tile<true>((ktiles - 1) * 32, q_idx, bS, Ebase, g, ln, wid, K, Vt,
                    qf, o, m_prev, l_sum, Sw, Sred);

    // --- normalize by row sums and store fp32 output
    const float invl = 1.f / l_sum;
    float inv[8];
#pragma unroll
    for (int r = 0; r < 8; ++r) inv[r] = bcast_lane(invl, r + 8 * g);
#pragma unroll
    for (int t = 0; t < 8; ++t) {
        const int n = Ebase + 16 * t + ln;
#pragma unroll
        for (int r = 0; r < 8; ++r) {
            const int m = q0 + r + 8 * g;
            out[((size_t)bS + m) * EDIM + n] = o[t][r] * inv[r];
        }
    }
}

// ---------------------------------------------------------------------------
// launch
// ---------------------------------------------------------------------------
extern "C" void kernel_launch(void* const* d_in, const int* in_sizes, int n_in,
                              void* d_out, int out_size, void* d_ws, size_t ws_size,
                              hipStream_t stream) {
    const float* xs   = (const float*)d_in[0];
    const float* WQ_w = (const float*)d_in[1];
    const float* WQ_b = (const float*)d_in[2];
    const float* WK_w = (const float*)d_in[3];
    const float* WK_b = (const float*)d_in[4];
    const float* WV_w = (const float*)d_in[5];
    const float* WV_b = (const float*)d_in[6];
    float* out = (float*)d_out;

    // workspace layout (bf16), all 16B-aligned slices
    bf16* Xbf = (bf16*)d_ws;                         // MTOT*HDIM
    bf16* Qbf = Xbf + (size_t)MTOT * HDIM;           // MTOT*EDIM
    bf16* Kbf = Qbf + (size_t)MTOT * EDIM;           // MTOT*EDIM
    bf16* Vtb = Kbf + (size_t)MTOT * EDIM;           // EDIM*MTOT (transposed)
    bf16* WQb = Vtb + (size_t)MTOT * EDIM;           // EDIM*HDIM
    bf16* WKb = WQb + (size_t)EDIM * HDIM;
    bf16* WVb = WKb + (size_t)EDIM * HDIM;

    cvt_f32_bf16<<<4096, 256, 0, stream>>>(xs,   Xbf, MTOT * HDIM);
    cvt_f32_bf16<<<1024, 256, 0, stream>>>(WQ_w, WQb, EDIM * HDIM);
    cvt_f32_bf16<<<1024, 256, 0, stream>>>(WK_w, WKb, EDIM * HDIM);
    cvt_f32_bf16<<<1024, 256, 0, stream>>>(WV_w, WVb, EDIM * HDIM);

    const float inv_sqrt_e = 0.03125f;  // 1/sqrt(1024), folded into Q

    // Q, K: A = X (rows = tokens), B = W (rows = E), out[m*EDIM + e], bias[e]
    dim3 pgQK(EDIM / 256, MTOT / 64);
    proj_gemm<false><<<pgQK, 256, 0, stream>>>(Xbf, WQb, WQ_b, Qbf, EDIM, inv_sqrt_e);
    proj_gemm<false><<<pgQK, 256, 0, stream>>>(Xbf, WKb, WK_b, Kbf, EDIM, 1.0f);

    // V^T: A = W (rows = E), B = X (rows = tokens), out[e*MTOT + m], bias[e]
    dim3 pgV(MTOT / 256, EDIM / 64);
    proj_gemm<true><<<pgV, 256, 0, stream>>>(WVb, Xbf, WV_b, Vtb, MTOT, 1.0f);

    dim3 fg(SEQ / 16, BATCH);
    flash_attn<<<fg, 256, 0, stream>>>(Qbf, Kbf, Vtb, out);
}
